// RelationalGraphConvolution_20023137534349
// MI455X (gfx1250) — matrix operand order repaired and checked by reading, not compiled, not run on hardware
//
#include <hip/hip_runtime.h>
#include <hip/hip_bf16.h>

// ---------------------------------------------------------------------------
// R-GCN layer for MI455X (gfx1250, wave32, WMMA).
//   out[b][o] = relu( x[b]@Wself^T + sum_r mean_masked(ne)[r][b]@Wrel[r]^T )
// Reformulated as a single f16 GEMM with K = (1+R)*128 = 2176:
//   A'[b][k]  = [ emb[nodes[b]] | agg[0][b] | ... | agg[15][b] ]   (f16)
//   W'[o][k]  = [ w_self[o]     | w_rel[0][o] | ... | w_rel[15][o] ] (f16)
//   out = relu(A' @ W'^T)   via v_wmma_f32_16x16x32_f16, f32 accum.
// ---------------------------------------------------------------------------

typedef _Float16 v16h  __attribute__((ext_vector_type(16)));
typedef _Float16 half8 __attribute__((ext_vector_type(8)));
typedef float    v8f   __attribute__((ext_vector_type(8)));

#define NB    4096          // batch of target nodes
#define SS    10            // sampled neighbors
#define RR    16            // relations
#define FF    128           // F_IN == F_OUT
#define KK    ((1 + RR) * FF)   // 2176 concatenated K dimension

// ---------------------------------------------------------------------------
// Kernel 1: gather + per-relation masked mean -> f16 A buffer [NB][KK]
// One block per target node b; 128 threads, thread t owns feature f = t.
// ---------------------------------------------------------------------------
__global__ void __launch_bounds__(128)
rgcn_agg_kernel(const float* __restrict__ emb,
                const int*   __restrict__ nodes,
                const int*   __restrict__ neighbors,
                const int*   __restrict__ rel_mask,
                _Float16*    __restrict__ Abuf) {
    const int b = blockIdx.x;
    const int f = threadIdx.x;

    __shared__ int   smask[SS * RR];   // 160 ints
    __shared__ float srcp[RR];         // 1/(count[r] + eps)

    for (int i = threadIdx.x; i < SS * RR; i += 128)
        smask[i] = rel_mask[(size_t)b * (SS * RR) + i];
    __syncthreads();

    if (threadIdx.x < RR) {
        float c = 0.0f;
        #pragma unroll
        for (int s = 0; s < SS; ++s) c += (float)smask[s * RR + threadIdx.x];
        srcp[threadIdx.x] = 1.0f / (c + 1e-10f);
    }
    __syncthreads();

    float sums[RR];
    #pragma unroll
    for (int r = 0; r < RR; ++r) sums[r] = 0.0f;

    for (int s = 0; s < SS; ++s) {
        const int nb = neighbors[(size_t)b * SS + s];
        const float v = emb[(size_t)nb * FF + f];   // coalesced 512B row
        #pragma unroll
        for (int r = 0; r < RR; ++r)
            if (smask[s * RR + r]) sums[r] += v;
    }

    _Float16* arow = Abuf + (size_t)b * KK;
    // self features into K-slot 0
    const int nd = nodes[b];
    arow[f] = (_Float16)emb[(size_t)nd * FF + f];
    // per-relation masked means into K-slots 1..16
    #pragma unroll
    for (int r = 0; r < RR; ++r)
        arow[(size_t)(1 + r) * FF + f] = (_Float16)(sums[r] * srcp[r]);
}

// ---------------------------------------------------------------------------
// Kernel 2: pack w_self [128][128] + w_rel [16][128][128] -> f16 W' [128][KK]
// ---------------------------------------------------------------------------
__global__ void __launch_bounds__(256)
rgcn_packw_kernel(const float* __restrict__ w_self,
                  const float* __restrict__ w_rel,
                  _Float16*    __restrict__ Wbuf) {
    const int i = blockIdx.x * 256 + threadIdx.x;
    if (i >= FF * KK) return;
    const int o = i / KK;
    const int k = i % KK;
    float v;
    if (k < FF) {
        v = w_self[(size_t)o * FF + k];
    } else {
        const int r = (k >> 7) - 1;     // k/128 - 1
        const int f = k & (FF - 1);
        v = w_rel[((size_t)r * FF + o) * FF + f];
    }
    Wbuf[i] = (_Float16)v;
}

// ---------------------------------------------------------------------------
// Kernel 3: GEMM  out[4096][128] = relu(A'[4096][2176] @ W'[128][2176]^T)
// 128 threads = 4 waves per block. Block owns M-tile of 16 rows; each wave
// owns a 16x32 output strip (2 f32 accumulators), A fragment reused twice.
// K loop: 68 steps of v_wmma_f32_16x16x32_f16.
// ---------------------------------------------------------------------------
__global__ void __launch_bounds__(128)
rgcn_gemm_kernel(const _Float16* __restrict__ Abuf,
                 const _Float16* __restrict__ Wbuf,
                 float*          __restrict__ out) {
    const int wave  = threadIdx.x >> 5;       // 0..3
    const int lane  = threadIdx.x & 31;
    const int halfw = lane >> 4;              // 0 or 1
    const int lid   = lane & 15;

    const int m_base = blockIdx.x * 16;       // 256 blocks -> 4096 rows
    const int n_base = wave * 32;             // 4 waves cover 128 cols

    const _Float16* pa  = Abuf + (size_t)(m_base + lid) * KK;
    const _Float16* pb0 = Wbuf + (size_t)(n_base + lid) * KK;
    const _Float16* pb1 = Wbuf + (size_t)(n_base + 16 + lid) * KK;

    v8f acc0 = {};
    v8f acc1 = {};

    for (int kt = 0; kt < KK; kt += 32) {
        // A fragment: 16 contiguous K-values of row (m_base+lid)
        half8 alo = *(const half8*)(pa + kt + halfw * 16);
        half8 ahi = *(const half8*)(pa + kt + halfw * 16 + 8);
        // B fragments: 16 contiguous K-values of column (n_base[+16]+lid)
        half8 b0l = *(const half8*)(pb0 + kt + halfw * 16);
        half8 b0h = *(const half8*)(pb0 + kt + halfw * 16 + 8);
        half8 b1l = *(const half8*)(pb1 + kt + halfw * 16);
        half8 b1h = *(const half8*)(pb1 + kt + halfw * 16 + 8);

        v16h a, b0, b1;
        #pragma unroll
        for (int i = 0; i < 8; ++i) {
            a[i] = alo[i];  a[8 + i] = ahi[i];
            b0[i] = b0l[i]; b0[8 + i] = b0h[i];
            b1[i] = b1l[i]; b1[8 + i] = b1h[i];
        }

        acc0 = __builtin_amdgcn_wmma_f32_16x16x32_f16(
                   false, a, false, b0, (short)0, acc0, false, false);
        acc1 = __builtin_amdgcn_wmma_f32_16x16x32_f16(
                   false, a, false, b1, (short)0, acc1, false, false);
    }

    // C/D layout: VGPR j -> row (halfw*8 + j), col lane%16. Fused ReLU, f32.
    float* po = out + (size_t)(m_base + halfw * 8) * FF;
    #pragma unroll
    for (int j = 0; j < 8; ++j) {
        float v0 = acc0[j];
        float v1 = acc1[j];
        po[(size_t)j * FF + n_base + lid]      = v0 > 0.0f ? v0 : 0.0f;
        po[(size_t)j * FF + n_base + 16 + lid] = v1 > 0.0f ? v1 : 0.0f;
    }
}

// ---------------------------------------------------------------------------
// Host launcher. Inputs (setup_inputs order):
//   0: emb [200000*128] f32     1: w_self [128*128] f32
//   2: w_rel [16*128*128] f32   3: nodes [4096] i32
//   4: neighbors [4096*10] i32  5: rel_mask [4096*10*16] i32
// Output: [4096*128] f32. Workspace: A' (17.8 MB f16) + W' (0.56 MB f16).
// ---------------------------------------------------------------------------
extern "C" void kernel_launch(void* const* d_in, const int* in_sizes, int n_in,
                              void* d_out, int out_size, void* d_ws, size_t ws_size,
                              hipStream_t stream) {
    const float* emb       = (const float*)d_in[0];
    const float* w_self    = (const float*)d_in[1];
    const float* w_rel     = (const float*)d_in[2];
    const int*   nodes     = (const int*)d_in[3];
    const int*   neighbors = (const int*)d_in[4];
    const int*   rel_mask  = (const int*)d_in[5];
    float*       out       = (float*)d_out;

    _Float16* Abuf = (_Float16*)d_ws;                    // [4096][2176]
    _Float16* Wbuf = Abuf + (size_t)NB * KK;             // [128][2176]

    rgcn_agg_kernel<<<NB, 128, 0, stream>>>(emb, nodes, neighbors, rel_mask, Abuf);

    const int wtot = FF * KK;
    rgcn_packw_kernel<<<(wtot + 255) / 256, 256, 0, stream>>>(w_self, w_rel, Wbuf);

    rgcn_gemm_kernel<<<NB / 16, 128, 0, stream>>>(Abuf, Wbuf, out);
}
